// MiMoV2FlashMoE_7679401525654
// MI455X (gfx1250) — compile-verified, matrix operand order checked by reading, not run on hardware
//
#include <hip/hip_runtime.h>

// Problem constants (match reference)
#define T_TOK 4096   // B*S tokens
#define HID   1024   // hidden size
#define IMD   768    // moe intermediate
#define NEXP  32     // experts
#define TOPK  4
#define TM    32     // tokens per MoE block tile

typedef __attribute__((ext_vector_type(16))) __bf16 v16bf;
typedef __attribute__((ext_vector_type(8)))  __bf16 v8bf;
typedef __attribute__((ext_vector_type(4)))  __bf16 v4bf;
typedef __attribute__((ext_vector_type(8)))  float  v8f;

// A-fragment (global bf16 activations or LDS h-tile): 16 bf16 per lane,
// two contiguous 16B chunks at +0 and +16 elements (K lo/hi interleave).
static __device__ __forceinline__ v16bf load_frag_bf16(const __bf16* p) {
  v8bf lo = *(const v8bf*)p;
  v8bf hi = *(const v8bf*)(p + 16);
  return __builtin_shufflevector(lo, hi,
         0,1,2,3,4,5,6,7,8,9,10,11,12,13,14,15);
}

// B-fragment from fp32 weights: 16 floats (two contiguous 32B chunks),
// converted to bf16 with native hardware converts.
static __device__ __forceinline__ v16bf load_frag_f32(const float* p) {
  float4 c0 = ((const float4*)p)[0];
  float4 c1 = ((const float4*)p)[1];
  float4 c2 = ((const float4*)(p + 16))[0];
  float4 c3 = ((const float4*)(p + 16))[1];
  v16bf r;
  r[0]  = (__bf16)c0.x; r[1]  = (__bf16)c0.y; r[2]  = (__bf16)c0.z; r[3]  = (__bf16)c0.w;
  r[4]  = (__bf16)c1.x; r[5]  = (__bf16)c1.y; r[6]  = (__bf16)c1.z; r[7]  = (__bf16)c1.w;
  r[8]  = (__bf16)c2.x; r[9]  = (__bf16)c2.y; r[10] = (__bf16)c2.z; r[11] = (__bf16)c2.w;
  r[12] = (__bf16)c3.x; r[13] = (__bf16)c3.y; r[14] = (__bf16)c3.z; r[15] = (__bf16)c3.w;
  return r;
}

#define WMMA_BF16(A, B, C) \
  __builtin_amdgcn_wmma_f32_16x16x32_bf16(false, (A), false, (B), (short)0, (C), false, false)

static __device__ __forceinline__ float fast_rcp(float x) {
  return __builtin_amdgcn_rcpf(x);            // v_rcp_f32 (TRANS, co-executes)
}

// ---------------- kernel 1: zero output + expert counters ----------------
__global__ void k_zero(float* __restrict__ y, int n, int* __restrict__ cnt, int ncnt) {
  int i = blockIdx.x * blockDim.x + threadIdx.x;
  int stride = gridDim.x * blockDim.x;
  for (int j = i; j < n; j += stride) y[j] = 0.0f;
  if (i < ncnt) cnt[i] = 0;
}

// ---------------- kernel 2: x fp32 -> bf16 ----------------
__global__ void k_cvt(const float* __restrict__ x, __bf16* __restrict__ xb) {
  int i = blockIdx.x * blockDim.x + threadIdx.x;        // [0, T*H/4)
  float4 v = ((const float4*)x)[i];
  v4bf o;
  o[0] = (__bf16)v.x; o[1] = (__bf16)v.y; o[2] = (__bf16)v.z; o[3] = (__bf16)v.w;
  ((v4bf*)xb)[i] = o;
}

// ---------------- kernel 3: router (one wave32 per token) ----------------
__global__ void k_router(const float* __restrict__ x, const float* __restrict__ wr,
                         int* __restrict__ cnt, int* __restrict__ rows,
                         float* __restrict__ wts) {
  int lane = threadIdx.x & 31;
  int wv   = threadIdx.x >> 5;
  int t = blockIdx.x * 8 + wv;

  const float4* xr  = (const float4*)(x  + (size_t)t    * HID);
  const float4* wrr = (const float4*)(wr + (size_t)lane * HID);
  float acc = 0.0f;
  for (int j = 0; j < HID / 4; ++j) {
    float4 a = xr[j], b = wrr[j];
    acc += a.x * b.x + a.y * b.y + a.z * b.z + a.w * b.w;
  }
  float cur = fast_rcp(1.0f + __expf(-acc));  // sigmoid score, lane = expert id

  float topw[TOPK]; int topi[TOPK]; float sum = 0.0f;
  #pragma unroll
  for (int k = 0; k < TOPK; ++k) {
    float bs = cur; int bi = lane;
    #pragma unroll
    for (int off = 16; off > 0; off >>= 1) {
      float os = __shfl_xor(bs, off, 32);
      int   oi = __shfl_xor(bi, off, 32);
      if (os > bs || (os == bs && oi < bi)) { bs = os; bi = oi; }
    }
    topw[k] = bs; topi[k] = bi; sum += bs;
    if (lane == bi) cur = -1e30f;             // knock winner out
  }
  float inv = fast_rcp(sum + 1e-20f);
  if (lane < TOPK) {
    int e = topi[lane];
    int slot = atomicAdd(&cnt[e], 1);
    rows[e * T_TOK + slot] = t;
    wts [e * T_TOK + slot] = topw[lane] * inv;
  }
}

// ---------------- kernel 4: fused SwiGLU expert MLP via WMMA ----------------
__global__ __launch_bounds__(256)
void k_moe(const __bf16* __restrict__ xb,
           const float* __restrict__ wg, const float* __restrict__ wu,
           const float* __restrict__ wd,
           const int* __restrict__ cnt, const int* __restrict__ rows,
           const float* __restrict__ wts, float* __restrict__ y) {
  __shared__ __bf16 hbuf[TM][IMD];            // bf16 silu(g)*u tile, 48 KB
  __shared__ int   stok[TM];
  __shared__ float swt [TM];

  int e = blockIdx.y;
  int ce = cnt[e];
  int base = blockIdx.x * TM;
  if (base >= ce) return;                     // uniform across block

  int tid = threadIdx.x;
  if (tid < TM) {
    int slot = base + tid;
    int cl = (slot < ce) ? slot : (ce - 1);
    stok[tid] = rows[e * T_TOK + cl];
    swt [tid] = (slot < ce) ? wts[e * T_TOK + slot] : 0.0f;   // pad -> weight 0
  }
  __syncthreads();

  int lane = tid & 31;
  int wv   = tid >> 5;
  int half = lane >> 4;       // 0: K lo half, 1: K hi half
  int l15  = lane & 15;
  int ksel = half * 8;

  // A-fragment row base pointers (bf16 activations, L2-resident 8MB)
  const __bf16* xr0 = xb + (size_t)stok[l15]      * HID + ksel;
  const __bf16* xr1 = xb + (size_t)stok[l15 + 16] * HID + ksel;

  const float* wge = wg + (size_t)e * IMD * HID;
  const float* wue = wu + (size_t)e * IMD * HID;

  // ---- phase 1: gate/up GEMMs + SwiGLU -> LDS (each wave owns 96 of I) ----
  for (int nt = 0; nt < 6; ++nt) {
    int n0 = wv * 96 + nt * 16;
    const float* bg = wge + (size_t)(n0 + l15) * HID + ksel;
    const float* bu = wue + (size_t)(n0 + l15) * HID + ksel;
    v8f ag0 = {0,0,0,0,0,0,0,0}, ag1 = {0,0,0,0,0,0,0,0};
    v8f au0 = {0,0,0,0,0,0,0,0}, au1 = {0,0,0,0,0,0,0,0};
    for (int k0 = 0; k0 < HID; k0 += 32) {
      v16bf a0 = load_frag_bf16(xr0 + k0);
      v16bf a1 = load_frag_bf16(xr1 + k0);
      v16bf fg = load_frag_f32(bg + k0);
      v16bf fu = load_frag_f32(bu + k0);
      ag0 = WMMA_BF16(a0, fg, ag0);
      ag1 = WMMA_BF16(a1, fg, ag1);
      au0 = WMMA_BF16(a0, fu, au0);
      au1 = WMMA_BF16(a1, fu, au1);
    }
    // D layout: lane<16 -> M=r, lane>=16 -> M=8+r; col = n0 + (lane&15)
    int m0 = half * 8;
    #pragma unroll
    for (int r = 0; r < 8; ++r) {
      float g0 = ag0[r], u0 = au0[r];
      float h0 = g0 * fast_rcp(1.0f + __expf(-g0)) * u0;   // silu(g)*u
      hbuf[m0 + r][n0 + l15] = (__bf16)h0;
      float g1 = ag1[r], u1 = au1[r];
      float h1 = g1 * fast_rcp(1.0f + __expf(-g1)) * u1;
      hbuf[m0 + r + 16][n0 + l15] = (__bf16)h1;
    }
  }
  __syncthreads();

  // ---- phase 2: down GEMM + weighted scatter (each wave owns 128 of H) ----
  const float* wde = wd + (size_t)e * HID * IMD;
  const __bf16* hp0 = &hbuf[l15][ksel];
  const __bf16* hp1 = &hbuf[l15 + 16][ksel];
  for (int nt = 0; nt < 8; ++nt) {
    int n0 = wv * 128 + nt * 16;
    const float* bd = wde + (size_t)(n0 + l15) * IMD + ksel;
    v8f ac0 = {0,0,0,0,0,0,0,0}, ac1 = {0,0,0,0,0,0,0,0};
    for (int k0 = 0; k0 < IMD; k0 += 32) {
      v16bf a0 = load_frag_bf16(hp0 + k0);     // from LDS (ds_load)
      v16bf a1 = load_frag_bf16(hp1 + k0);
      v16bf fb = load_frag_f32(bd + k0);
      ac0 = WMMA_BF16(a0, fb, ac0);
      ac1 = WMMA_BF16(a1, fb, ac1);
    }
    int m0 = half * 8;
    #pragma unroll
    for (int r = 0; r < 8; ++r) {
      int ma = m0 + r;
      atomicAdd(&y[(size_t)stok[ma] * HID + (n0 + l15)], ac0[r] * swt[ma]);
      int mb = ma + 16;
      atomicAdd(&y[(size_t)stok[mb] * HID + (n0 + l15)], ac1[r] * swt[mb]);
    }
  }
}

extern "C" void kernel_launch(void* const* d_in, const int* in_sizes, int n_in,
                              void* d_out, int out_size, void* d_ws, size_t ws_size,
                              hipStream_t stream) {
  (void)in_sizes; (void)n_in; (void)out_size; (void)ws_size;
  const float* x  = (const float*)d_in[0];   // [T, H]
  const float* wr = (const float*)d_in[1];   // [E, H]
  const float* wg = (const float*)d_in[2];   // [E, I, H]
  const float* wu = (const float*)d_in[3];   // [E, I, H]
  const float* wd = (const float*)d_in[4];   // [E, H, I]
  float* y = (float*)d_out;                  // [T, H]

  // workspace layout
  char* ws = (char*)d_ws;
  __bf16* xbf = (__bf16*)ws;                                 // T*H bf16 = 8 MB
  size_t off = (size_t)T_TOK * HID * 2;
  int*   cnt  = (int*)(ws + off);  off += (size_t)NEXP * 4;
  int*   rows = (int*)(ws + off);  off += (size_t)NEXP * T_TOK * 4;
  float* wts  = (float*)(ws + off);

  k_zero  <<<4096, 256, 0, stream>>>(y, T_TOK * HID, cnt, NEXP);
  k_cvt   <<<(T_TOK * HID / 4) / 256, 256, 0, stream>>>(x, xbf);
  k_router<<<T_TOK / 8, 256, 0, stream>>>(x, wr, cnt, rows, wts);
  k_moe   <<<dim3(T_TOK / TM, NEXP), 256, 0, stream>>>(xbf, wg, wu, wd,
                                                       cnt, rows, wts, y);
}
